// ScaledDotProductAttention_4999341932926
// MI455X (gfx1250) — matrix operand order repaired and checked by reading, not compiled, not run on hardware
//
#include <hip/hip_runtime.h>

typedef _Float16 v16h __attribute__((ext_vector_type(16)));
typedef _Float16 v8h  __attribute__((ext_vector_type(8)));
typedef _Float16 v4h  __attribute__((ext_vector_type(4)));
typedef float    v8f  __attribute__((ext_vector_type(8)));

#define SEQ   2048
#define DIM   64
#define BLK_N 64
#define KPAD  72   // LDS row stride (halves); 144B rows keep b64 stores 8B-aligned

// 1/sqrt(64) * log2(e): softmax runs in exp2 domain
#define QSCALE 0.1803368801111204f

__device__ __forceinline__ v16h cat8(v8h lo, v8h hi) {
    return __builtin_shufflevector(lo, hi, 0,1,2,3,4,5,6,7,8,9,10,11,12,13,14,15);
}

// DPP16 row_ror:N move within each 16-lane row (no LDS, no waits)
template <int N>
__device__ __forceinline__ float dpp_ror(float x) {
    int r = __builtin_amdgcn_update_dpp(0, __float_as_int(x),
                                        0x120 | N, 0xF, 0xF, true);
    return __int_as_float(r);
}

__device__ __forceinline__ float rowmax16(float x) {
    x = fmaxf(x, dpp_ror<1>(x));
    x = fmaxf(x, dpp_ror<2>(x));
    x = fmaxf(x, dpp_ror<4>(x));
    x = fmaxf(x, dpp_ror<8>(x));
    return x;   // every lane in the 16-lane row holds the row max
}

__device__ __forceinline__ float rowsum16(float x) {
    x += dpp_ror<1>(x);
    x += dpp_ror<2>(x);
    x += dpp_ror<4>(x);
    x += dpp_ror<8>(x);
    return x;
}

__global__ __launch_bounds__(128)
void fa_fwd_kernel(const float* __restrict__ Qg,
                   const float* __restrict__ Kg,
                   const float* __restrict__ Vg,
                   float* __restrict__ Og) {
    __shared__ __align__(32) _Float16 Klds [64 * KPAD];   // K block, row-major [kv][d]
    __shared__ __align__(32) _Float16 Vtlds[64 * KPAD];   // V block, transposed [d][kv]
    __shared__ __align__(32) _Float16 Plds [4 * 16 * 64]; // per-wave P tile [16][64]

    const int tid  = threadIdx.x;
    const int wave = tid >> 5;
    const int lane = tid & 31;
    const int hi   = lane >> 4;   // half-wave select
    const int ln   = lane & 15;   // position within half-wave

    const int b     = blockIdx.x >> 5;   // batch
    const int qtile = blockIdx.x & 31;   // 64-row q tile index

    const long qbase = (long)b * SEQ * DIM + (long)(qtile * 64 + wave * 16) * DIM;

    // ---- Q fragments in WMMA A layout, pre-scaled by QSCALE ----
    v16h qf[2];
    {
        const int m = ln;                  // A row = lane&15
        #pragma unroll
        for (int kc = 0; kc < 2; ++kc) {   // two K=32 chunks over d=64
            const float* p0 = Qg + qbase + (long)m * DIM + kc * 32 + 8 * hi;
            float4 a0 = ((const float4*)p0)[0];
            float4 a1 = ((const float4*)p0)[1];
            float4 b0 = ((const float4*)(p0 + 16))[0];
            float4 b1 = ((const float4*)(p0 + 16))[1];
            float t[16] = {a0.x,a0.y,a0.z,a0.w, a1.x,a1.y,a1.z,a1.w,
                           b0.x,b0.y,b0.z,b0.w, b1.x,b1.y,b1.z,b1.w};
            v16h f;
            #pragma unroll
            for (int j = 0; j < 16; ++j) f[j] = (_Float16)(t[j] * QSCALE);
            qf[kc] = f;
        }
    }

    // ---- online-softmax state (each lane tracks its 8 rows 8*hi+r) ----
    float mprev[8], lsum[8];
    v8f   o[4];
    #pragma unroll
    for (int r = 0; r < 8; ++r) { mprev[r] = -INFINITY; lsum[r] = 0.f; }
    #pragma unroll
    for (int dt = 0; dt < 4; ++dt)
        #pragma unroll
        for (int r = 0; r < 8; ++r) o[dt][r] = 0.f;

    _Float16* Pw = Plds + wave * 16 * 64;

    for (int blk = 0; blk < SEQ / BLK_N; ++blk) {
        const long kbase = (long)b * SEQ * DIM + (long)blk * BLK_N * DIM;

        __syncthreads();                       // protect LDS from previous iter

        // ---- stage K (wide loads, packed b64 stores) ----
        #pragma unroll
        for (int i = 0; i < 8; ++i) {          // 1024 4-half chunks / 128 threads
            int id = tid + i * 128;
            int r  = id >> 4;                  // 0..63 (kv row)
            int cg = (id & 15) * 4;            // 0..60 (d group)
            float4 f = *(const float4*)(Kg + kbase + (long)r * DIM + cg);
            v4h h4 = {(_Float16)f.x, (_Float16)f.y, (_Float16)f.z, (_Float16)f.w};
            *(v4h*)(Klds + r * KPAD + cg) = h4;
        }
        // ---- stage V transposed (coalesced loads, packed b64 stores) ----
        #pragma unroll
        for (int i = 0; i < 8; ++i) {
            int id = tid + i * 128;
            int c  = id & 63;                  // d column
            int rg = (id >> 6) * 4;            // 0..60 (kv row group)
            float f0 = Vg[kbase + (long)(rg + 0) * DIM + c];
            float f1 = Vg[kbase + (long)(rg + 1) * DIM + c];
            float f2 = Vg[kbase + (long)(rg + 2) * DIM + c];
            float f3 = Vg[kbase + (long)(rg + 3) * DIM + c];
            v4h h4 = {(_Float16)f0, (_Float16)f1, (_Float16)f2, (_Float16)f3};
            *(v4h*)(Vtlds + c * KPAD + rg) = h4;
        }
        __syncthreads();

        // prefetch next KV block rows into cache while we compute
        if (blk + 1 < SEQ / BLK_N) {
            const float* nxt = (tid < 64)
                ? (Kg + kbase + BLK_N * DIM + (long)tid * DIM)
                : (Vg + kbase + BLK_N * DIM + (long)(tid - 64) * DIM);
            __builtin_prefetch(nxt, 0, 1);
        }

        // ---- S = (Q*QSCALE) * K^T : 4 N-tiles x 2 K-chunks ----
        v8f s[4];
        #pragma unroll
        for (int t = 0; t < 4; ++t) {
            v8f acc;
            #pragma unroll
            for (int r = 0; r < 8; ++r) acc[r] = 0.f;
            #pragma unroll
            for (int kc = 0; kc < 2; ++kc) {
                const _Float16* bp = Klds + (t * 16 + ln) * KPAD + kc * 32 + 8 * hi;
                v16h bf = cat8(*(const v8h*)bp, *(const v8h*)(bp + 16));
                acc = __builtin_amdgcn_wmma_f32_16x16x32_f16(
                        false, qf[kc], false, bf, (short)0, acc, false, false);
            }
            s[t] = acc;
        }

        // ---- online softmax (exp2 domain) over the 64-wide block ----
        #pragma unroll
        for (int r = 0; r < 8; ++r) {
            float mx = fmaxf(fmaxf(s[0][r], s[1][r]), fmaxf(s[2][r], s[3][r]));
            mx = rowmax16(mx);                      // DPP all-reduce, 16 lanes
            float mn    = fmaxf(mprev[r], mx);
            float alpha = exp2f(mprev[r] - mn);
            mprev[r] = mn;

            float sum = 0.f;
            #pragma unroll
            for (int t = 0; t < 4; ++t) {
                float p = exp2f(s[t][r] - mn);
                s[t][r] = p;
                sum += p;
            }
            sum = rowsum16(sum);                    // DPP all-reduce, 16 lanes
            lsum[r] = lsum[r] * alpha + sum;

            #pragma unroll
            for (int dt = 0; dt < 4; ++dt) o[dt][r] *= alpha;

            // P row (8*hi + r), columns t*16 + ln
            #pragma unroll
            for (int t = 0; t < 4; ++t)
                Pw[(8 * hi + r) * 64 + t * 16 + ln] = (_Float16)s[t][r];
        }

        // LDS is in-order within a wave; stop compiler from reordering DS ops
        asm volatile("" ::: "memory");

        // ---- O += P * V : A from per-wave P tile, B from Vt ----
        #pragma unroll
        for (int kc = 0; kc < 2; ++kc) {
            const _Float16* ap = Pw + ln * 64 + kc * 32;
            v16h pa = cat8(*(const v8h*)(ap + 8 * hi),
                           *(const v8h*)(ap + 16 + 8 * hi));
            #pragma unroll
            for (int dt = 0; dt < 4; ++dt) {
                const _Float16* vb = Vtlds + (dt * 16 + ln) * KPAD + kc * 32 + 8 * hi;
                v16h bf = cat8(*(const v8h*)vb, *(const v8h*)(vb + 16));
                o[dt] = __builtin_amdgcn_wmma_f32_16x16x32_f16(
                          false, pa, false, bf, (short)0, o[dt], false, false);
            }
        }
    }

    // ---- normalize and store ----
    #pragma unroll
    for (int r = 0; r < 8; ++r) {
        float inv = 1.0f / lsum[r];
        #pragma unroll
        for (int dt = 0; dt < 4; ++dt)
            Og[qbase + (long)(8 * hi + r) * DIM + dt * 16 + ln] = o[dt][r] * inv;
    }
}

extern "C" void kernel_launch(void* const* d_in, const int* in_sizes, int n_in,
                              void* d_out, int out_size, void* d_ws, size_t ws_size,
                              hipStream_t stream) {
    const float* q = (const float*)d_in[0];
    const float* k = (const float*)d_in[1];
    const float* v = (const float*)d_in[2];
    float* out = (float*)d_out;
    (void)in_sizes; (void)n_in; (void)out_size; (void)d_ws; (void)ws_size;

    dim3 grid(32 * 32);   // 32 batches x 32 q-tiles of 64 rows
    dim3 block(128);      // 4 wave32 waves, one 16-row WMMA stripe each
    fa_fwd_kernel<<<grid, block, 0, stream>>>(q, k, v, out);
}
